// ChaosSNN_47072841564535
// MI455X (gfx1250) — matrix operand order repaired and checked by reading, not compiled, not run on hardware
//
#include <hip/hip_runtime.h>

typedef __attribute__((ext_vector_type(16))) __bf16     v16bf;
typedef __attribute__((ext_vector_type(8)))  float      v8f;
typedef __attribute__((ext_vector_type(4)))  unsigned   u32x4;

union FragU {
    v16bf bf;
    u32x4 u[2];
};

// Problem constants
constexpr int  Bdim = 2048;   // batch rows (M)
constexpr int  Rdim = 4096;   // reduction (K)
constexpr int  Odim = 8192;   // output cols (N)

// GEMM tiling
constexpr int BM = 128, BN = 128, BK = 32;
constexpr int NK = Rdim / BK;   // 128 K-steps

__device__ __forceinline__ unsigned short f32_to_bf16(float f) {
    unsigned u = __float_as_uint(f);
    // round-to-nearest-even
    u += 0x7FFFu + ((u >> 16) & 1u);
    return (unsigned short)(u >> 16);
}
__device__ __forceinline__ float bf16_to_f32(unsigned short h) {
    return __uint_as_float(((unsigned)h) << 16);
}

// CDNA5 async global->LDS copy (ASYNCcnt tracked, no VGPR staging).
// vdst = LDS byte offset, vaddr = 64-bit global address, SADDR = off.
__device__ __forceinline__ void async_copy_b128(unsigned ldsOff,
                                                const void* gaddr) {
    asm volatile("global_load_async_to_lds_b128 %0, %1, off"
                 :
                 : "v"(ldsOff), "v"(gaddr)
                 : "memory");
}
__device__ __forceinline__ void wait_async0() {
    asm volatile("s_wait_asynccnt 0" ::: "memory");
}
__device__ __forceinline__ unsigned lds_off(const void* p) {
    // Flat pointer into the LDS aperture: addr[31:0] is the LDS byte offset.
    return (unsigned)(unsigned long long)p;
}

// ---------------------------------------------------------------------------
// Stage 1: per-block partial sums of |w| (deterministic; no atomics)
// ---------------------------------------------------------------------------
__global__ __launch_bounds__(256)
void absSumKernel(const float* __restrict__ w, float* __restrict__ partials) {
    constexpr long N4 = (long)Odim * Rdim / 4;   // 8,388,608 float4
    const float4* w4 = (const float4*)w;
    float s = 0.f;
    for (long i = (long)blockIdx.x * 256 + threadIdx.x; i < N4;
         i += (long)gridDim.x * 256) {
        float4 v = w4[i];
        s += fabsf(v.x) + fabsf(v.y) + fabsf(v.z) + fabsf(v.w);
    }
    __shared__ float red[256];
    red[threadIdx.x] = s;
    __syncthreads();
    for (int off = 128; off > 0; off >>= 1) {
        if (threadIdx.x < off) red[threadIdx.x] += red[threadIdx.x + off];
        __syncthreads();
    }
    if (threadIdx.x == 0) partials[blockIdx.x] = red[0];
}

// ---------------------------------------------------------------------------
// Stage 2: fold 2048 partials (fixed order) -> threshold = 0.7*mean(|w|)
// ---------------------------------------------------------------------------
__global__ __launch_bounds__(256)
void finalizeKernel(float* __restrict__ ws) {
    float s = 0.f;
    for (int i = threadIdx.x; i < 2048; i += 256) s += ws[i];
    __shared__ float red[256];
    red[threadIdx.x] = s;
    __syncthreads();
    for (int off = 128; off > 0; off >>= 1) {
        if (threadIdx.x < off) red[threadIdx.x] += red[threadIdx.x + off];
        __syncthreads();
    }
    if (threadIdx.x == 0)
        ws[2048] = 0.7f * red[0] / (float)((long)Odim * Rdim);
}

// ---------------------------------------------------------------------------
// Stage 3: ternary-quantize weight to bf16 in {-1, 0, +1}
// ---------------------------------------------------------------------------
__global__ __launch_bounds__(256)
void quantKernel(const float* __restrict__ w, const float* __restrict__ thr,
                 unsigned short* __restrict__ wq) {
    const float t = thr[0];
    const long base = ((long)blockIdx.x * 256 + threadIdx.x) * 8;
    const float4* w4 = (const float4*)(w + base);
    float4 a = w4[0], b = w4[1];
    float x[8] = {a.x, a.y, a.z, a.w, b.x, b.y, b.z, b.w};
    unsigned q[8];
#pragma unroll
    for (int i = 0; i < 8; ++i) {
        unsigned v = 0u;
        if (fabsf(x[i]) > t) v = (x[i] > 0.f) ? 0x3F80u : 0xBF80u;
        q[i] = v;
    }
    u32x4 p;
    p[0] = q[0] | (q[1] << 16);
    p[1] = q[2] | (q[3] << 16);
    p[2] = q[4] | (q[5] << 16);
    p[3] = q[6] | (q[7] << 16);
    *(u32x4*)(wq + base) = p;
}

// ---------------------------------------------------------------------------
// Stage 4: split noise f32 -> bf16 hi + bf16 lo (hi+lo ~ exact f32)
// ---------------------------------------------------------------------------
__global__ __launch_bounds__(256)
void splitKernel(const float* __restrict__ noise,
                 unsigned short* __restrict__ hi,
                 unsigned short* __restrict__ lo) {
    const long base = ((long)blockIdx.x * 256 + threadIdx.x) * 8;
    const float4* n4 = (const float4*)(noise + base);
    float4 a = n4[0], b = n4[1];
    float x[8] = {a.x, a.y, a.z, a.w, b.x, b.y, b.z, b.w};
    unsigned h[8], l[8];
#pragma unroll
    for (int i = 0; i < 8; ++i) {
        unsigned short hb = f32_to_bf16(x[i]);
        float hf = bf16_to_f32(hb);
        unsigned short lb = f32_to_bf16(x[i] - hf);
        h[i] = hb;
        l[i] = lb;
    }
    u32x4 ph, pl;
    ph[0] = h[0] | (h[1] << 16); ph[1] = h[2] | (h[3] << 16);
    ph[2] = h[4] | (h[5] << 16); ph[3] = h[6] | (h[7] << 16);
    pl[0] = l[0] | (l[1] << 16); pl[1] = l[2] | (l[3] << 16);
    pl[2] = l[4] | (l[5] << 16); pl[3] = l[6] | (l[7] << 16);
    *(u32x4*)(hi + base) = ph;
    *(u32x4*)(lo + base) = pl;
}

// ---------------------------------------------------------------------------
// Stage 5: WMMA GEMM with double-buffered LDS fed by async global->LDS.
// out[m,n] = sigma * sum_k (Ahi+Alo)[m,k] * Wq[n,k]
// 128x128 block tile, BK=32, 8 waves (2 in M x 4 in N), each wave 64x32.
// v_wmma_f32_16x16x32_bf16, hi + lo issues share the B fragment.
// ---------------------------------------------------------------------------
__global__ __launch_bounds__(256)
void gemmKernel(const unsigned short* __restrict__ Ahi,
                const unsigned short* __restrict__ Alo,
                const unsigned short* __restrict__ Wq,
                const float* __restrict__ sigmaPtr,
                float* __restrict__ out) {
    __shared__ __align__(16) unsigned short sAhi[2][BM * BK];
    __shared__ __align__(16) unsigned short sAlo[2][BM * BK];
    __shared__ __align__(16) unsigned short sW[2][BN * BK];

    const int tid = threadIdx.x;
    const int n0 = blockIdx.x * BN;
    const int m0 = blockIdx.y * BM;

    const int wid = tid >> 5;
    const int lane = tid & 31;
    const int wm = wid & 1;      // wave M group: 0..1 (64 rows each)
    const int wn = wid >> 1;     // wave N group: 0..3 (32 cols each)
    const int lrow = lane & 15;  // M (A) or N (B) within 16
    const int khalf = lane >> 4; // K half selector

    // staging: thread -> (row 0..127, 16-element half of a 32-wide K row)
    const int srow = tid >> 1;
    const int scol = (tid & 1) * 16;

    // per-thread global base pointers for staging (advance by k0 elements)
    const unsigned short* gA = Ahi + (size_t)(m0 + srow) * Rdim + scol;
    const unsigned short* gL = Alo + (size_t)(m0 + srow) * Rdim + scol;
    const unsigned short* gW = Wq + (size_t)(n0 + srow) * Rdim + scol;

    // per-thread LDS byte offsets of this thread's staging slot, per buffer
    unsigned dA[2], dL[2], dW[2];
#pragma unroll
    for (int b = 0; b < 2; ++b) {
        dA[b] = lds_off(&sAhi[b][srow * BK + scol]);
        dL[b] = lds_off(&sAlo[b][srow * BK + scol]);
        dW[b] = lds_off(&sW[b][srow * BK + scol]);
    }

    v8f acc[4][2];
    v8f zero = {};
#pragma unroll
    for (int mt = 0; mt < 4; ++mt)
#pragma unroll
        for (int nt = 0; nt < 2; ++nt) acc[mt][nt] = zero;

    // prologue: stage tile 0 into buffer 0
    {
        async_copy_b128(dA[0], gA);
        async_copy_b128(dA[0] + 16, gA + 8);
        async_copy_b128(dL[0], gL);
        async_copy_b128(dL[0] + 16, gL + 8);
        async_copy_b128(dW[0], gW);
        async_copy_b128(dW[0] + 16, gW + 8);
    }

    for (int kt = 0; kt < NK; ++kt) {
        const int cur = kt & 1;
        const int nxt = cur ^ 1;

        // current buffer's async stage (the only in-flight ops) -> done,
        // then make it visible to all waves. This barrier also proves every
        // wave finished reading buffer `nxt` in iteration kt-1, so it is
        // safe to refill it below.
        wait_async0();
        __syncthreads();

        if (kt + 1 < NK) {
            const int k1 = (kt + 1) * BK;
            async_copy_b128(dA[nxt], gA + k1);
            async_copy_b128(dA[nxt] + 16, gA + k1 + 8);
            async_copy_b128(dL[nxt], gL + k1);
            async_copy_b128(dL[nxt] + 16, gL + k1 + 8);
            async_copy_b128(dW[nxt], gW + k1);
            async_copy_b128(dW[nxt] + 16, gW + k1 + 8);
        }

        // A fragments (16-bit A 16x32 layout: lane m=lrow, K-half=khalf;
        // elems 0..7 -> k = 8*khalf.., elems 8..15 -> k = 16+8*khalf..)
        FragU aHi[4], aLo[4], bFr[2];
#pragma unroll
        for (int mt = 0; mt < 4; ++mt) {
            const int row = wm * 64 + mt * 16 + lrow;
            const u32x4* bh = (const u32x4*)(&sAhi[cur][row * BK]);
            aHi[mt].u[0] = bh[khalf];
            aHi[mt].u[1] = bh[2 + khalf];
            const u32x4* bl = (const u32x4*)(&sAlo[cur][row * BK]);
            aLo[mt].u[0] = bl[khalf];
            aLo[mt].u[1] = bl[2 + khalf];
        }
        // B fragments (16-bit B 32x16: lane n=lrow, K-half=khalf;
        // elems 0..7 -> k = 8*khalf.., elems 8..15 -> k = 16+8*khalf..)
#pragma unroll
        for (int nt = 0; nt < 2; ++nt) {
            const int row = wn * 32 + nt * 16 + lrow;
            const u32x4* bb = (const u32x4*)(&sW[cur][row * BK]);
            bFr[nt].u[0] = bb[khalf];
            bFr[nt].u[1] = bb[2 + khalf];
        }

#pragma unroll
        for (int mt = 0; mt < 4; ++mt)
#pragma unroll
            for (int nt = 0; nt < 2; ++nt) {
                acc[mt][nt] = __builtin_amdgcn_wmma_f32_16x16x32_bf16(
                    false, aHi[mt].bf, false, bFr[nt].bf, (short)0,
                    acc[mt][nt], false, false);
                acc[mt][nt] = __builtin_amdgcn_wmma_f32_16x16x32_bf16(
                    false, aLo[mt].bf, false, bFr[nt].bf, (short)0,
                    acc[mt][nt], false, false);
            }
    }

    const float sigma = sigmaPtr[0];
#pragma unroll
    for (int mt = 0; mt < 4; ++mt)
#pragma unroll
        for (int nt = 0; nt < 2; ++nt)
#pragma unroll
            for (int v = 0; v < 8; ++v) {
                const int m = m0 + wm * 64 + mt * 16 + v + 8 * khalf;
                const int n = n0 + wn * 32 + nt * 16 + lrow;
                out[(size_t)m * Odim + n] = acc[mt][nt][v] * sigma;
            }
}

// ---------------------------------------------------------------------------
// Host-side launcher
// ---------------------------------------------------------------------------
extern "C" void kernel_launch(void* const* d_in, const int* in_sizes, int n_in,
                              void* d_out, int out_size, void* d_ws,
                              size_t ws_size, hipStream_t stream) {
    const float* noise  = (const float*)d_in[0];   // [2048, 4096]
    const float* weight = (const float*)d_in[1];   // [8192, 4096]
    const float* sigma  = (const float*)d_in[2];   // scalar
    float* out = (float*)d_out;                    // [2048, 8192]

    char* ws = (char*)d_ws;
    float* partials = (float*)ws;                          // 2048 floats
    float* thr = partials + 2048;                          // 1 float
    unsigned short* AhiP = (unsigned short*)(ws + 65536);  // 16 MiB
    unsigned short* AloP = AhiP + (size_t)Bdim * Rdim;     // 16 MiB
    unsigned short* WqP  = AloP + (size_t)Bdim * Rdim;     // 64 MiB

    absSumKernel<<<2048, 256, 0, stream>>>(weight, partials);
    finalizeKernel<<<1, 256, 0, stream>>>(partials);
    quantKernel<<<(Odim * (long)Rdim / 8) / 256, 256, 0, stream>>>(weight, thr,
                                                                   WqP);
    splitKernel<<<((long)Bdim * Rdim / 8) / 256, 256, 0, stream>>>(noise, AhiP,
                                                                   AloP);
    dim3 grid(Odim / BN, Bdim / BM);  // (64, 16)
    gemmKernel<<<grid, 256, 0, stream>>>(AhiP, AloP, WqP, sigma, out);
}
